// SelfSupervisedContrastiveLoss2_62113817035248
// MI455X (gfx1250) — compile-verified, matrix operand order checked by reading, not run on hardware
//
#include <hip/hip_runtime.h>
#include <hip/hip_bf16.h>

// ---------------------------------------------------------------------------
// Fused InfoNCE-style contrastive loss for MI455X (gfx1250, wave32, WMMA).
// Q = [fn; pn] (8192x1024), K = [fn; pn; nn] (12288x1024) in bf16.
// Z[a] = sum_j exp(2 * q_a . k_j); loss assembled from Z, diagonals, pos.
//
// expsum kernel: A tile (128 rows x 1024) resident in LDS for the whole key
// loop; B staged through a double-buffered LDS chunk via async global->LDS
// copies (ASYNCcnt) shared by all 8 waves.  292 KB LDS of the WGP's 320 KB.
// Fragments are double-buffered in registers and pinned with sched_barrier
// so the DS loads of K-step ks+1 stay in flight across the WMMAs of step ks.
// ---------------------------------------------------------------------------

#define B_ROWS   4096
#define N_ROWS   4096
#define DIM      1024
#define NQ       (2 * B_ROWS)        // 8192 anchors
#define NK       (3 * B_ROWS)        // 12288 keys
#define INV_TAU  2.0f
#define EPS      1e-8f
#define KEY_SPLITS 8

#define LDA 1032                     // padded A row stride (ushorts): banks 4l mod 64
#define LDB 136                      // padded B row stride (ushorts)
#define KEYTILE 64                   // keys per tile
#define KCHUNK  128                  // K elements staged per chunk

typedef __attribute__((ext_vector_type(16))) __bf16   v16bf;
typedef __attribute__((ext_vector_type(8)))  float    v8f;
typedef __attribute__((ext_vector_type(4)))  unsigned int v4u;
typedef __attribute__((ext_vector_type(2)))  unsigned int v2u;

union FragBf {
    v16bf v;
    v4u   q[2];
};

#if defined(__has_builtin)
#if __has_builtin(__builtin_amdgcn_global_load_async_to_lds_b128)
#define HAVE_ASYNC_LDS 1
#endif
#if __has_builtin(__builtin_amdgcn_s_wait_asynccnt)
#define HAVE_WAIT_ASYNC 1
#endif
#if __has_builtin(__builtin_amdgcn_sched_barrier)
#define HAVE_SCHED_BARRIER 1
#endif
#endif

// GCC-style int4 vector, matching the builtin's parameter pointee type:
//   '__attribute__((__vector_size__(4 * sizeof(int)))) int __device__ *'
typedef int async_b128_t __attribute__((vector_size(16)));
typedef __attribute__((address_space(1))) async_b128_t* gptr_b128;
typedef __attribute__((address_space(3))) async_b128_t* lptr_b128;

__device__ __forceinline__ void async_copy16(const unsigned short* g,
                                             unsigned short* l) {
#if defined(HAVE_ASYNC_LDS)
    __builtin_amdgcn_global_load_async_to_lds_b128(
        (gptr_b128)(g), (lptr_b128)(l), 0, 0);
#else
    *(v4u*)l = *(const v4u*)g;
#endif
}

__device__ __forceinline__ void wait_async_all() {
#if defined(HAVE_ASYNC_LDS)
#if defined(HAVE_WAIT_ASYNC)
    __builtin_amdgcn_s_wait_asynccnt(0);
#else
    asm volatile("s_wait_asynccnt 0x0" ::: "memory");
#endif
#endif
}

__device__ __forceinline__ void sched_fence() {
#if defined(HAVE_SCHED_BARRIER)
    __builtin_amdgcn_sched_barrier(0);
#endif
}

__device__ __forceinline__ unsigned short f2bf_rne(float x) {
    unsigned int u = __float_as_uint(x);
    unsigned int r = (u + 0x7FFFu + ((u >> 16) & 1u)) >> 16;
    return (unsigned short)r;
}
__device__ __forceinline__ float bf2f(unsigned short u) {
    return __uint_as_float(((unsigned int)u) << 16);
}

// ---------------------------------------------------------------------------
// Kernel 0: zero the Z accumulators and the output scalar.
// ---------------------------------------------------------------------------
__global__ void init_kernel(float* __restrict__ Z, float* __restrict__ out) {
    int i = blockIdx.x * blockDim.x + threadIdx.x;
    if (i < NQ) Z[i] = 0.0f;
    if (i == 0) out[0] = 0.0f;
}

// ---------------------------------------------------------------------------
// Kernel 1: L2-normalize each of the 12288 rows, emit bf16 key matrix.
// ---------------------------------------------------------------------------
__global__ void normalize_kernel(const float* __restrict__ f,
                                 const float* __restrict__ fpos,
                                 const float* __restrict__ fneg,
                                 unsigned short* __restrict__ keys) {
    const int row  = blockIdx.x;
    const int t    = threadIdx.x;
    const int lane = t & 31;
    const int wave = t >> 5;

    const float* src;
    if (row < B_ROWS)            src = f    + (size_t)row * DIM;
    else if (row < 2 * B_ROWS)   src = fpos + (size_t)(row - B_ROWS) * DIM;
    else                         src = fneg + (size_t)(row - 2 * B_ROWS) * DIM;

    float4 x = *(const float4*)(src + t * 4);
    float s = x.x * x.x + x.y * x.y + x.z * x.z + x.w * x.w;

    #pragma unroll
    for (int m = 1; m < 32; m <<= 1) s += __shfl_xor(s, m);

    __shared__ float red[8];
    if (lane == 0) red[wave] = s;
    __syncthreads();
    float total = red[0] + red[1] + red[2] + red[3] +
                  red[4] + red[5] + red[6] + red[7];

    float scale = 1.0f / fmaxf(sqrtf(total), EPS);

    unsigned int lo = ((unsigned int)f2bf_rne(x.y * scale) << 16) |
                       (unsigned int)f2bf_rne(x.x * scale);
    unsigned int hi = ((unsigned int)f2bf_rne(x.w * scale) << 16) |
                       (unsigned int)f2bf_rne(x.z * scale);
    v2u packed = {lo, hi};
    *(v2u*)(keys + (size_t)row * DIM + t * 4) = packed;
}

// ---------------------------------------------------------------------------
// Kernel 2: pos[i] = 2 * cos_sim(f_i, fpos_i) in full fp32.
// ---------------------------------------------------------------------------
__global__ void pos_kernel(const float* __restrict__ f,
                           const float* __restrict__ fpos,
                           float* __restrict__ pos) {
    const int i    = blockIdx.x;
    const int t    = threadIdx.x;
    const int lane = t & 31;
    const int wave = t >> 5;

    float4 a = *(const float4*)(f    + (size_t)i * DIM + t * 4);
    float4 b = *(const float4*)(fpos + (size_t)i * DIM + t * 4);

    float d  = a.x * b.x + a.y * b.y + a.z * b.z + a.w * b.w;
    float na = a.x * a.x + a.y * a.y + a.z * a.z + a.w * a.w;
    float nb = b.x * b.x + b.y * b.y + b.z * b.z + b.w * b.w;

    #pragma unroll
    for (int m = 1; m < 32; m <<= 1) {
        d  += __shfl_xor(d,  m);
        na += __shfl_xor(na, m);
        nb += __shfl_xor(nb, m);
    }

    __shared__ float rd[8], rna[8], rnb[8];
    if (lane == 0) { rd[wave] = d; rna[wave] = na; rnb[wave] = nb; }
    __syncthreads();
    if (t == 0) {
        float D = 0, NA = 0, NB = 0;
        #pragma unroll
        for (int w = 0; w < 8; ++w) { D += rd[w]; NA += rna[w]; NB += rnb[w]; }
        pos[i] = INV_TAU * D / (fmaxf(sqrtf(NA), EPS) * fmaxf(sqrtf(NB), EPS));
    }
}

// ---------------------------------------------------------------------------
// Kernel 3: fused exp-sum GEMM, LDS-resident A + double-buffered async B.
// 256 threads = 8 waves.  Wave tile: 16 anchors x 64 keys (4 acc chains).
// A-frag (16-bit A 16x32): lanes 0-15 row M=l, K in [0,8)+[16,24);
// lanes 16-31 row M=l, K in [8,16)+[24,32).  B mirrors with lane = column.
// C/D: element r = row (r + 8*(lane>>4)), col (lane&15).
// ---------------------------------------------------------------------------
__device__ __forceinline__ void load_frags(const unsigned short* aBase,
                                           const unsigned short* bbuf,
                                           int kb, int ks, int l15, int half,
                                           FragBf& a, FragBf* b) {
    const unsigned short* ap = aBase + kb + ks * 32;
    a.q[0] = *(const v4u*)ap;
    a.q[1] = *(const v4u*)(ap + 16);
    #pragma unroll
    for (int t = 0; t < 4; ++t) {
        const unsigned short* bp =
            bbuf + (t * 16 + l15) * LDB + ks * 32 + half * 8;
        b[t].q[0] = *(const v4u*)bp;
        b[t].q[1] = *(const v4u*)(bp + 16);
    }
}

__global__ void __launch_bounds__(256)
expsum_kernel(const unsigned short* __restrict__ keys,
              float* __restrict__ Z,
              int keysPerSplit) {
    __shared__ unsigned short ldsA[128 * LDA];        // 258 KB
    __shared__ unsigned short ldsB[2][KEYTILE * LDB]; // 34 KB

    const int tid  = threadIdx.x;
    const int lane = tid & 31;
    const int wave = tid >> 5;
    const int half = lane >> 4;
    const int l15  = lane & 15;

    const int rowTile  = blockIdx.x * 128;                 // anchor rows
    const int keyStart = blockIdx.y * keysPerSplit;

    // ---- stage the full A tile (128 x 1024) into LDS, async ----
    #pragma unroll 8
    for (int i = 0; i < 64; ++i) {
        int idx = tid + i * 256;            // 16384 chunks of 8 ushorts
        int row = idx >> 7;                 // 0..127
        int kc  = (idx & 127) * 8;          // 0..1016
        async_copy16(keys + (size_t)(rowTile + row) * DIM + kc,
                     &ldsA[row * LDA + kc]);
    }

    // ---- stage B chunk 0 of key tile 0 ----
    {
        #pragma unroll
        for (int i = 0; i < 4; ++i) {
            int idx = tid + i * 256;        // 1024 chunks
            int row = idx >> 4;             // 0..63
            int kc  = (idx & 15) * 8;       // 0..120
            async_copy16(keys + (size_t)(keyStart + row) * DIM + kc,
                         &ldsB[0][row * LDB + kc]);
        }
    }

    const unsigned short* aBase = &ldsA[(wave * 16 + l15) * LDA + half * 8];

    float rs[8];
    #pragma unroll
    for (int r = 0; r < 8; ++r) rs[r] = 0.0f;

    v8f acc[4];
    #pragma unroll
    for (int t = 0; t < 4; ++t)
        acc[t] = (v8f){0.f, 0.f, 0.f, 0.f, 0.f, 0.f, 0.f, 0.f};

    const int numTiles    = keysPerSplit / KEYTILE;  // 24
    const int totalChunks = numTiles * 8;            // 8 chunks of K=128 per tile

    for (int cg = 0; cg < totalChunks; ++cg) {
        const int c = cg & 7;                        // K-chunk within tile

        wait_async_all();
        __syncthreads();                             // chunk cg staged; prev reads done

        // stage next chunk into the other buffer (overlaps with compute below)
        if (cg + 1 < totalChunks) {
            const int cg1     = cg + 1;
            const int keyBase = keyStart + (cg1 >> 3) * KEYTILE;
            const int kBase   = (cg1 & 7) * KCHUNK;
            unsigned short* bbuf = ldsB[cg1 & 1];
            #pragma unroll
            for (int i = 0; i < 4; ++i) {
                int idx = tid + i * 256;
                int row = idx >> 4;
                int kc  = (idx & 15) * 8;
                async_copy16(keys + (size_t)(keyBase + row) * DIM + kBase + kc,
                             &bbuf[row * LDB + kc]);
            }
        }

        // ---- compute: 4 K-steps of 32, 4 key fragments each.
        //      Fragments double-buffered in registers; sched_barrier pins
        //      next-step DS loads above this step's WMMAs so they overlap. ----
        const unsigned short* bbuf = ldsB[cg & 1];
        const int kb = c * KCHUNK;

        FragBf aF[2];
        FragBf bF[2][4];
        load_frags(aBase, bbuf, kb, 0, l15, half, aF[0], bF[0]);

        #pragma unroll
        for (int ks = 0; ks < 4; ++ks) {
            const int cur = ks & 1;
            if (ks < 3)
                load_frags(aBase, bbuf, kb, ks + 1, l15, half,
                           aF[cur ^ 1], bF[cur ^ 1]);
            sched_fence();   // DS loads of step ks+1 must not sink below here
            #pragma unroll
            for (int t = 0; t < 4; ++t)
                acc[t] = __builtin_amdgcn_wmma_f32_16x16x32_bf16(
                    false, aF[cur].v, false, bF[cur][t].v,
                    (short)0, acc[t], false, false);
        }

        // ---- end of a key tile: exp + accumulate row sums, reset accs ----
        if (c == 7) {
            #pragma unroll
            for (int t = 0; t < 4; ++t) {
                #pragma unroll
                for (int r = 0; r < 8; ++r)
                    rs[r] += __expf(INV_TAU * acc[t][r]);
                acc[t] = (v8f){0.f, 0.f, 0.f, 0.f, 0.f, 0.f, 0.f, 0.f};
            }
        }
    }

    // reduce over the 16 columns held in each half-wave
    #pragma unroll
    for (int r = 0; r < 8; ++r) {
        float v = rs[r];
        v += __shfl_xor(v, 1);
        v += __shfl_xor(v, 2);
        v += __shfl_xor(v, 4);
        v += __shfl_xor(v, 8);
        rs[r] = v;
    }
    if (l15 == 0) {
        #pragma unroll
        for (int r = 0; r < 8; ++r)
            atomicAdd(&Z[rowTile + wave * 16 + half * 8 + r], rs[r]);
    }
}

// ---------------------------------------------------------------------------
// Kernel 4: per-anchor finalize.  One wave per anchor.
// ---------------------------------------------------------------------------
__global__ void finalize_kernel(const unsigned short* __restrict__ keys,
                                const float* __restrict__ Z,
                                const float* __restrict__ pos,
                                float* __restrict__ out) {
    const int lane = threadIdx.x & 31;
    const int wave = threadIdx.x >> 5;
    const int a    = blockIdx.x * 8 + wave;

    const unsigned short* row = keys + (size_t)a * DIM;
    float s = 0.0f;
    #pragma unroll
    for (int it = 0; it < 8; ++it) {
        int k = it * 128 + lane * 4;
        v2u p = *(const v2u*)(row + k);
        float x0 = bf2f((unsigned short)(p.x & 0xFFFFu));
        float x1 = bf2f((unsigned short)(p.x >> 16));
        float x2 = bf2f((unsigned short)(p.y & 0xFFFFu));
        float x3 = bf2f((unsigned short)(p.y >> 16));
        s += x0 * x0 + x1 * x1 + x2 * x2 + x3 * x3;
    }
    #pragma unroll
    for (int m = 1; m < 32; m <<= 1) s += __shfl_xor(s, m);

    if (lane == 0) {
        float denom = Z[a] - __expf(INV_TAU * s);
        float term  = 0.5f * __logf(denom);
        if (a < B_ROWS) term -= pos[a];
        atomicAdd(out, term * (1.0f / (float)B_ROWS));
    }
}

// ---------------------------------------------------------------------------
// Launch
// ---------------------------------------------------------------------------
extern "C" void kernel_launch(void* const* d_in, const int* in_sizes, int n_in,
                              void* d_out, int out_size, void* d_ws, size_t ws_size,
                              hipStream_t stream) {
    const float* f    = (const float*)d_in[0];
    const float* fpos = (const float*)d_in[1];
    const float* fneg = (const float*)d_in[2];
    float* out = (float*)d_out;

    const size_t keysBytes = (size_t)NK * DIM * sizeof(unsigned short); // 24 MB
    unsigned short* keys = (unsigned short*)d_ws;
    float* Z   = (float*)((char*)d_ws + keysBytes);
    float* pos = Z + NQ;

    init_kernel<<<(NQ + 255) / 256, 256, 0, stream>>>(Z, out);
    normalize_kernel<<<NK, 256, 0, stream>>>(f, fpos, fneg, keys);
    pos_kernel<<<B_ROWS, 256, 0, stream>>>(f, fpos, pos);

    dim3 grid(NQ / 128, KEY_SPLITS);
    expsum_kernel<<<grid, 256, 0, stream>>>(keys, Z, NK / KEY_SPLITS);

    finalize_kernel<<<NQ / 8, 256, 0, stream>>>(keys, Z, pos, out);
}
